// MoeRouter_9019431322100
// MI455X (gfx1250) — matrix operand order repaired and checked by reading, not compiled, not run on hardware
//
#include <hip/hip_runtime.h>
#include <hip/hip_bf16.h>

typedef __attribute__((ext_vector_type(16))) _Float16 v16h;
typedef __attribute__((ext_vector_type(8)))  _Float16 v8h;
typedef __attribute__((ext_vector_type(8)))  float    v8f;
typedef __attribute__((ext_vector_type(4)))  float    v4f;

#define D_DIM   2048
#define E_EXP   64
#define KC      64                 // k-chunk of W staged per LDS buffer (floats)
#define NCHUNK  (D_DIM / KC)       // 32
#define LWS     72                 // LDS row stride in halves (bank-conflict-free b128)

__device__ __forceinline__ bool better(float v, int i, float w, int j) {
    // jax.lax.top_k ordering: larger value first; ties -> lower index first
    return (v > w) || ((v == w) && (i < j));
}

__global__ __launch_bounds__(128, 1)
void moe_router_fused(const float* __restrict__ x,
                      const float* __restrict__ W,
                      float* __restrict__ out_w,
                      int* __restrict__ out_i,
                      int T)
{
    __shared__ _Float16 lw[2][E_EXP * LWS];   // 18.4 KB, double-buffered W slice (f16)

    const int tid    = threadIdx.x;
    const int lane   = tid & 31;
    const int wave   = tid >> 5;              // 0..3
    const int laneLo = lane & 15;
    const int hi     = lane >> 4;             // 0: lanes 0-15, 1: lanes 16-31

    // ---- cooperative W staging: 128 threads cover 64 rows x 64 floats (32 each)
    const int se = tid >> 1;                  // expert row 0..63
    const int sq = tid & 1;                   // half-row: 32 contiguous floats
    const float* wsrc = W + (size_t)se * D_DIM + sq * 32;

    // ---- this wave's 16 token rows
    const int rowBase = blockIdx.x * 64 + wave * 16;
    const float* xrow = x + (size_t)(rowBase + laneLo) * D_DIM;

    v8f acc[4] = {v8f{}, v8f{}, v8f{}, v8f{}};   // 4 expert tiles of 16

    // helper: cvt 8 floats (2 v4f) -> 8 halves and store 16B to LDS
    auto cvt_store8 = [](_Float16* dst, v4f f0, v4f f1) {
        v8h h;
#pragma unroll
        for (int j = 0; j < 4; ++j) { h[j] = (_Float16)f0[j]; h[4 + j] = (_Float16)f1[j]; }
        *(v8h*)dst = h;
    };

    // ---- prologue: fetch + convert + store chunk 0
    {
        v4f f[8];
#pragma unroll
        for (int q = 0; q < 8; ++q) f[q] = *(const v4f*)(wsrc + q * 4);
        _Float16* dst = &lw[0][se * LWS + sq * 32];
#pragma unroll
        for (int q = 0; q < 4; ++q) cvt_store8(dst + q * 8, f[2 * q], f[2 * q + 1]);
    }

    for (int c = 0; c < NCHUNK; ++c) {
        __syncthreads();                       // chunk c resident in buf c&1
        const int buf = c & 1;

        // issue next chunk's global W fetch early (latency overlaps compute)
        v4f g[8];
        if (c + 1 < NCHUNK) {
            const float* nw = wsrc + (c + 1) * KC;
#pragma unroll
            for (int q = 0; q < 8; ++q) g[q] = *(const v4f*)(nw + q * 4);
        }

        // ---- A fragments for both K=32 steps (global NT stream of x)
        v16h A[2];
#pragma unroll
        for (int s = 0; s < 2; ++s) {
            const int kb = c * KC + s * 32;
            // h[0..7] = x[row, kb+off .. +7], h[8..15] = x[row, kb+16+off ..]
            const v4f* ap = (const v4f*)(xrow + kb + hi * 8);
            v4f a0 = __builtin_nontemporal_load(ap + 0);
            v4f a1 = __builtin_nontemporal_load(ap + 1);
            v4f a2 = __builtin_nontemporal_load(ap + 4);
            v4f a3 = __builtin_nontemporal_load(ap + 5);
#pragma unroll
            for (int j = 0; j < 4; ++j) {
                A[s][j]      = (_Float16)a0[j];
                A[s][4 + j]  = (_Float16)a1[j];
                A[s][8 + j]  = (_Float16)a2[j];
                A[s][12 + j] = (_Float16)a3[j];
            }
        }

        // ---- 8 WMMA steps; B fragments double-buffered from LDS
        const _Float16* bwave = &lw[buf][laneLo * LWS + hi * 8];
        auto loadB = [&](int idx) -> v16h {    // idx = s*4 + t
            const int s = idx >> 2, t = idx & 3;
            const _Float16* bp = bwave + s * 32 + t * 16 * LWS;
            v8h blo = *(const v8h*)bp;
            v8h bhi = *(const v8h*)(bp + 16);
            return __builtin_shufflevector(blo, bhi,
                     0,1,2,3,4,5,6,7,8,9,10,11,12,13,14,15);
        };

        v16h B0 = loadB(0);
        v16h B1 = loadB(1);
#pragma unroll
        for (int idx = 0; idx < 8; ++idx) {
            v16h Bn;
            if (idx + 2 < 8) Bn = loadB(idx + 2);
            const int s = idx >> 2, t = idx & 3;
            acc[t] = __builtin_amdgcn_wmma_f32_16x16x32_f16(
                         false, A[s], false, B0, (short)0, acc[t], false, false);
            B0 = B1;
            if (idx + 2 < 8) B1 = Bn;
        }

        // convert + store next chunk into the other buffer (after WMMAs, before barrier)
        if (c + 1 < NCHUNK) {
            _Float16* dst = &lw[buf ^ 1][se * LWS + sq * 32];
#pragma unroll
            for (int q = 0; q < 4; ++q) cvt_store8(dst + q * 8, g[2 * q], g[2 * q + 1]);
        }

        // ---- explicit scheduler pipeline for this region:
        //  VMEM reads first, 2 B-frags of DS reads ahead, then WMMA interleaved
        //  with the next B-frag's DS reads; DS writes (staging) last.
        __builtin_amdgcn_sched_group_barrier(0x020, 16, 0);   // global reads (A + W prefetch)
        __builtin_amdgcn_sched_group_barrier(0x100, 4, 0);    // DS read: B frags 0,1
#pragma unroll
        for (int i = 0; i < 6; ++i) {
            __builtin_amdgcn_sched_group_barrier(0x008, 1, 0); // 1 WMMA
            __builtin_amdgcn_sched_group_barrier(0x100, 2, 0); // DS read: B frag i+2
        }
        __builtin_amdgcn_sched_group_barrier(0x008, 2, 0);    // last 2 WMMAs
        __builtin_amdgcn_sched_group_barrier(0x200, 4, 0);    // DS write: staging stores
    }

    // ---- fused softmax + top-2 epilogue.
    // C layout: VGPR r, lanes0-15 -> token r, lanes16-31 -> token r+8; N = laneLo.
#pragma unroll
    for (int r = 0; r < 8; ++r) {
        float b0v = acc[0][r]; int b0i = laneLo;        // expert 0*16+laneLo
        float b1v = -3.402823466e38f; int b1i = 0x7fffffff;
#pragma unroll
        for (int t = 1; t < 4; ++t) {
            float v = acc[t][r]; int i = t * 16 + laneLo;
            if (better(v, i, b0v, b0i)) { b1v = b0v; b1i = b0i; b0v = v; b0i = i; }
            else if (better(v, i, b1v, b1i)) { b1v = v; b1i = i; }
        }
        // 16-lane butterfly merge of sorted top-2 pairs (stays within wave halves)
#pragma unroll
        for (int m = 1; m <= 8; m <<= 1) {
            float o0v = __shfl_xor(b0v, m, 32);
            int   o0i = __shfl_xor(b0i, m, 32);
            float o1v = __shfl_xor(b1v, m, 32);
            int   o1i = __shfl_xor(b1i, m, 32);
            if (better(o0v, o0i, b0v, b0i)) {
                float nv; int ni;
                if (better(b0v, b0i, o1v, o1i)) { nv = b0v; ni = b0i; }
                else                            { nv = o1v; ni = o1i; }
                b0v = o0v; b0i = o0i; b1v = nv; b1i = ni;
            } else if (better(o0v, o0i, b1v, b1i)) {
                b1v = o0v; b1i = o0i;
            }
        }
        // full softmax denominator Z = sum exp(l - max); max == b0v (broadcast by butterfly)
        float z = 0.0f;
#pragma unroll
        for (int t = 0; t < 4; ++t) z += __expf(acc[t][r] - b0v);
#pragma unroll
        for (int m = 1; m <= 8; m <<= 1) z += __shfl_xor(z, m, 32);

        if (laneLo == 0) {
            int token = rowBase + r + hi * 8;
            float e1 = __expf(b1v - b0v);                 // e0 == 1
            float den = 1.0f + e1 + 1e-9f * z;            // (p0+p1+eps)/p_scale
            out_w[token * 2 + 0] = 1.0f / den;
            out_w[token * 2 + 1] = e1 / den;
            out_i[token * 2 + 0] = b0i;
            out_i[token * 2 + 1] = b1i;
        }
    }
}

extern "C" void kernel_launch(void* const* d_in, const int* in_sizes, int n_in,
                              void* d_out, int out_size, void* d_ws, size_t ws_size,
                              hipStream_t stream) {
    const float* x = (const float*)d_in[0];
    const float* W = (const float*)d_in[1];
    const int T = in_sizes[0] / D_DIM;        // 16384
    float* out_w = (float*)d_out;             // [T,2] f32
    int*   out_i = (int*)d_out + 2 * T;       // [T,2] i32 (concatenated flat)
    dim3 grid(T / 64), block(128);            // 4 waves x 16 tokens per block
    moe_router_fused<<<grid, block, 0, stream>>>(x, W, out_w, out_i, T);
}
